// YatSphericalCausalAttention_56367150793414
// MI455X (gfx1250) — matrix operand order repaired and checked by reading, not compiled, not run on hardware
//
#include <hip/hip_runtime.h>

// ---------------------------------------------------------------------------
// Yat spherical causal attention for gfx1250 (MI455X), wave32 + WMMA bf16.
//   k1: qkv = x @ qkv_w^T + b ; head split; L2-normalize q,k; pack bf16
//   k2: streaming causal attention (single pass; no softmax), WMMA for
//       S = Qn Kn^T and O += P V; async global->LDS staging of Q/K tiles
//   k3: out = attn @ out_w^T + out_b (f32 output)
// ---------------------------------------------------------------------------

typedef __attribute__((ext_vector_type(16))) __bf16 v16bf;
typedef __attribute__((ext_vector_type(8)))  __bf16 v8bf;
typedef __attribute__((ext_vector_type(8)))  float  v8f;

constexpr int D  = 1024;
constexpr int H  = 16;
constexpr int HD = 64;
constexpr int BB = 2;
constexpr int T  = 2048;
constexpr int NROW = BB * T;              // 4096
constexpr float CCONST = 2.0f + 1e-6f;

static __device__ __forceinline__ v8f wmma_bf16(v16bf a, v16bf b, v8f c) {
  return __builtin_amdgcn_wmma_f32_16x16x32_bf16(false, a, false, b,
                                                 (short)0, c, false, false);
}

// Async global->LDS copy, 16 bytes per lane (CDNA5 ASYNCcnt path).
static __device__ __forceinline__ void async_load_b128(unsigned ldsOff,
                                                       const void* gptr) {
  unsigned long long ga = (unsigned long long)(uintptr_t)gptr;
  asm volatile("global_load_async_to_lds_b128 %0, %1, off"
               :
               : "v"(ldsOff), "v"(ga)
               : "memory");
}
static __device__ __forceinline__ void wait_asynccnt0() {
  asm volatile("s_wait_asynccnt 0x0" ::: "memory");
}

// A fragment (16x32 bf16): lane 0-15 -> row m, K {ko..ko+7, ko+16..ko+23},
// ko = 0 (lanes 0-15) or 8 (lanes 16-31).  Two 16B LDS reads per lane.
static __device__ __forceinline__ v16bf
load_afrag(const __bf16* p, int stride, int mBase, int kBase, int lane) {
  const int m  = mBase + (lane & 15);
  const int ko = (lane & 16) ? 8 : 0;
  const __bf16* row = p + m * stride + kBase + ko;
  const v8bf lo = *(const v8bf*)(row);
  const v8bf hi = *(const v8bf*)(row + 16);
  return __builtin_shufflevector(lo, hi, 0, 1, 2, 3, 4, 5, 6, 7,
                                 8, 9, 10, 11, 12, 13, 14, 15);
}

// B fragment (32x16 bf16): lane 0-15 -> col n, K {ko..ko+7, ko+16..ko+23},
// ko = 0 (lanes 0-15) or 8 (lanes 16-31).  Storage is n-major: B[k][n] at p[n*stride + k].  Two 16B reads.
static __device__ __forceinline__ v16bf
load_bfrag(const __bf16* p, int stride, int nBase, int kBase, int lane) {
  const int n  = nBase + (lane & 15);
  const int ko = (lane & 16) ? 8 : 0;
  const __bf16* row = p + n * stride + kBase + ko;
  const v8bf lo = *(const v8bf*)(row);
  const v8bf hi = *(const v8bf*)(row + 16);
  return __builtin_shufflevector(lo, hi, 0, 1, 2, 3, 4, 5, 6, 7,
                                 8, 9, 10, 11, 12, 13, 14, 15);
}

// ---------------------------------------------------------------------------
// Kernel 1: QKV projection + head split + L2 norm of q,k + bf16 pack.
// Block tile 128(M) x 64(N), 8 waves x (16x64).  N tile == exactly one head.
// ---------------------------------------------------------------------------
__global__ __launch_bounds__(256)
void qkv_kernel(const float* __restrict__ x, const float* __restrict__ w,
                const float* __restrict__ bias,
                __bf16* __restrict__ Qn, __bf16* __restrict__ Kn,
                __bf16* __restrict__ Vb) {
  constexpr int LDA = 40, LDB = 40;  // 80-byte rows: 16B-aligned fragments
  __shared__ __align__(16) __bf16 As[128 * LDA];
  __shared__ __align__(16) __bf16 Bs[64 * LDB];
  const int tid  = threadIdx.x;
  const int lane = tid & 31;
  const int wv   = tid >> 5;
  const int mBlk = blockIdx.x * 128;
  const int nBlk = blockIdx.y * 64;

  v8f acc[4] = {};

  for (int k0 = 0; k0 < D; k0 += 32) {
    __syncthreads();
    for (int i = tid; i < 128 * 8; i += 256) {  // A: 128x32 f32 -> bf16
      const int r = i >> 3, c4 = (i & 7) * 4;
      const float4 v = *(const float4*)(x + (size_t)(mBlk + r) * D + k0 + c4);
      __bf16* dst = &As[r * LDA + c4];
      dst[0] = (__bf16)v.x; dst[1] = (__bf16)v.y;
      dst[2] = (__bf16)v.z; dst[3] = (__bf16)v.w;
    }
    for (int i = tid; i < 64 * 8; i += 256) {   // B: 64x32 of qkv_w, n-major
      const int r = i >> 3, c4 = (i & 7) * 4;
      const float4 v = *(const float4*)(w + (size_t)(nBlk + r) * D + k0 + c4);
      __bf16* dst = &Bs[r * LDB + c4];
      dst[0] = (__bf16)v.x; dst[1] = (__bf16)v.y;
      dst[2] = (__bf16)v.z; dst[3] = (__bf16)v.w;
    }
    __syncthreads();
    const v16bf a = load_afrag(As, LDA, wv * 16, 0, lane);
#pragma unroll
    for (int t = 0; t < 4; ++t) {
      const v16bf b = load_bfrag(Bs, LDB, t * 16, 0, lane);
      acc[t] = wmma_bf16(a, b, acc[t]);
    }
  }

  const int rhalf = (lane & 16) ? 8 : 0;
#pragma unroll
  for (int t = 0; t < 4; ++t) {
    const float bv = bias[nBlk + t * 16 + (lane & 15)];
#pragma unroll
    for (int r = 0; r < 8; ++r) acc[t][r] += bv;
  }
  const int sec = nBlk >> 10;  // 0=q, 1=k, 2=v
  if (sec < 2) {
#pragma unroll
    for (int r = 0; r < 8; ++r) {
      float ss = 0.f;
#pragma unroll
      for (int t = 0; t < 4; ++t) ss += acc[t][r] * acc[t][r];
      ss += __shfl_xor(ss, 1); ss += __shfl_xor(ss, 2);
      ss += __shfl_xor(ss, 4); ss += __shfl_xor(ss, 8);
      const float scl = 1.0f / fmaxf(sqrtf(ss), 1e-12f);
#pragma unroll
      for (int t = 0; t < 4; ++t) acc[t][r] *= scl;
    }
  }
  const int h = (nBlk & 1023) >> 6;
  __bf16* dst = (sec == 0) ? Qn : (sec == 1) ? Kn : Vb;
#pragma unroll
  for (int t = 0; t < 4; ++t) {
    const int dd = t * 16 + (lane & 15);
#pragma unroll
    for (int r = 0; r < 8; ++r) {
      const int m = mBlk + wv * 16 + rhalf + r;
      const int bi = m >> 11, ti = m & (T - 1);
      dst[(((size_t)bi * H + h) * T + ti) * HD + dd] = (__bf16)acc[t][r];
    }
  }
}

// ---------------------------------------------------------------------------
// Kernel 2: streaming causal Yat attention.  Block = (b, h, 128 q rows),
// 8 waves x 16 q rows.  Q/K tiles staged with async global->LDS b128; V
// staged transposed so all B fragments are pairs of ds_load_b128.
// ---------------------------------------------------------------------------
__global__ __launch_bounds__(256)
void attn_kernel(const __bf16* __restrict__ Qn, const __bf16* __restrict__ Kn,
                 const __bf16* __restrict__ Vb, float* __restrict__ attn) {
  constexpr int LDV = 72;  // 144-byte rows, 16B-aligned
  __shared__ __align__(16) __bf16 Qs[128 * HD];
  __shared__ __align__(16) __bf16 Ks[64 * HD];
  __shared__ __align__(16) __bf16 Vt[HD * LDV];    // transposed: [hd][kv]
  __shared__ __align__(16) __bf16 Ps[8][16 * HD];  // per-wave P scratch

  const int tid  = threadIdx.x;
  const int lane = tid & 31;
  const int wv   = tid >> 5;
  const int qBlk = blockIdx.x * 128;
  const int h    = blockIdx.y;
  const int b    = blockIdx.z;
  const size_t headOff = (((size_t)b * H + h) * T) * HD;
  const __bf16* qbase = Qn + headOff;
  const __bf16* kbase = Kn + headOff;
  const __bf16* vbase = Vb + headOff;

  // Q tile 128x64 bf16: async copy straight into LDS.
  {
    const unsigned qsOff = (unsigned)(uintptr_t)(&Qs[0]);
    const char* src = (const char*)(qbase + (size_t)qBlk * HD);
    for (int i = tid; i < 128 * 8; i += 256)
      async_load_b128(qsOff + (unsigned)i * 16u, src + (size_t)i * 16);
    wait_asynccnt0();
  }
  __syncthreads();

  v16bf qf[2];
  qf[0] = load_afrag(Qs, HD, wv * 16, 0, lane);
  qf[1] = load_afrag(Qs, HD, wv * 16, 32, lane);

  v8f o[4] = {};
  float rowsum[8] = {0, 0, 0, 0, 0, 0, 0, 0};
  const int rhalf   = (lane & 16) ? 8 : 0;
  const int qRowMax = qBlk + wv * 16 + 15;
  const unsigned ksOff = (unsigned)(uintptr_t)(&Ks[0]);

  for (int kv0 = 0; kv0 < qBlk + 128; kv0 += 64) {
    __syncthreads();
    {
      // K tile 64x64: async copy (row layout is already fragment-friendly).
      const char* ksrc = (const char*)(kbase + (size_t)kv0 * HD);
      for (int i = tid; i < 64 * 8; i += 256)
        async_load_b128(ksOff + (unsigned)i * 16u, ksrc + (size_t)i * 16);
      // V tile 64x64: load + transpose into Vt[hd][kv].
      for (int i = tid; i < 64 * 8; i += 256) {
        const int r = i >> 3, c8 = (i & 7) * 8;
        const uint4 pv =
            *(const uint4*)(vbase + (size_t)(kv0 + r) * HD + c8);
        const __bf16* e = (const __bf16*)&pv;
#pragma unroll
        for (int j = 0; j < 8; ++j) Vt[(c8 + j) * LDV + r] = e[j];
      }
      // Prefetch next K tile into cache while this one is consumed.
      if (kv0 + 64 < qBlk + 128)
        __builtin_prefetch(kbase + (size_t)(kv0 + 64) * HD, 0, 0);
      wait_asynccnt0();
    }
    __syncthreads();
    if (kv0 > qRowMax) continue;  // fully masked for this wave

    __bf16* P = Ps[wv];
    // S = Qn Kn^T (16x64 per wave), then yat score transform + causal mask.
#pragma unroll
    for (int ct = 0; ct < 4; ++ct) {
      v8f s = {};
      s = wmma_bf16(qf[0], load_bfrag(Ks, HD, ct * 16, 0, lane), s);
      s = wmma_bf16(qf[1], load_bfrag(Ks, HD, ct * 16, 32, lane), s);
      const int gcol = kv0 + ct * 16 + (lane & 15);
#pragma unroll
      for (int r = 0; r < 8; ++r) {
        const int grow = qBlk + wv * 16 + rhalf + r;
        const float xd  = s[r];
        const float den = fmaxf(CCONST - 2.0f * xd, 1e-6f);
        const float sc  = (gcol <= grow) ? (xd * xd) / den : 0.0f;
        rowsum[r] += sc;
        P[(rhalf + r) * HD + ct * 16 + (lane & 15)] = (__bf16)sc;
      }
    }
    __builtin_amdgcn_wave_barrier();  // order P stores vs cross-lane reloads
    // O += P @ V (contraction over 64 kv, two 32-K steps; Vt is n-major).
#pragma unroll
    for (int kk = 0; kk < 2; ++kk) {
      const v16bf pf = load_afrag(P, HD, 0, kk * 32, lane);
#pragma unroll
      for (int nt = 0; nt < 4; ++nt)
        o[nt] = wmma_bf16(pf, load_bfrag(Vt, LDV, nt * 16, kk * 32, lane),
                          o[nt]);
    }
  }

  // Normalize rows by (sum + 1e-6); store f32 in [B,T,D] layout.
#pragma unroll
  for (int r = 0; r < 8; ++r) {
    float ssum = rowsum[r];
    ssum += __shfl_xor(ssum, 1); ssum += __shfl_xor(ssum, 2);
    ssum += __shfl_xor(ssum, 4); ssum += __shfl_xor(ssum, 8);
    const float inv = 1.0f / (ssum + 1e-6f);
    const int grow = qBlk + wv * 16 + rhalf + r;
    float* orow = attn + ((size_t)b * T + grow) * D + h * HD;
#pragma unroll
    for (int nt = 0; nt < 4; ++nt)
      orow[nt * 16 + (lane & 15)] = o[nt][r] * inv;
  }
}

// ---------------------------------------------------------------------------
// Kernel 3: output projection, out = attn @ out_w^T + out_b (f32 out).
// ---------------------------------------------------------------------------
__global__ __launch_bounds__(256)
void proj_kernel(const float* __restrict__ a_in, const float* __restrict__ w,
                 const float* __restrict__ bias, float* __restrict__ out) {
  constexpr int LDA = 40, LDB = 40;
  __shared__ __align__(16) __bf16 As[128 * LDA];
  __shared__ __align__(16) __bf16 Bs[64 * LDB];
  const int tid  = threadIdx.x;
  const int lane = tid & 31;
  const int wv   = tid >> 5;
  const int mBlk = blockIdx.x * 128;
  const int nBlk = blockIdx.y * 64;

  v8f acc[4] = {};

  for (int k0 = 0; k0 < D; k0 += 32) {
    __syncthreads();
    for (int i = tid; i < 128 * 8; i += 256) {
      const int r = i >> 3, c4 = (i & 7) * 4;
      const float4 v = *(const float4*)(a_in + (size_t)(mBlk + r) * D + k0 + c4);
      __bf16* dst = &As[r * LDA + c4];
      dst[0] = (__bf16)v.x; dst[1] = (__bf16)v.y;
      dst[2] = (__bf16)v.z; dst[3] = (__bf16)v.w;
    }
    for (int i = tid; i < 64 * 8; i += 256) {
      const int r = i >> 3, c4 = (i & 7) * 4;
      const float4 v = *(const float4*)(w + (size_t)(nBlk + r) * D + k0 + c4);
      __bf16* dst = &Bs[r * LDB + c4];
      dst[0] = (__bf16)v.x; dst[1] = (__bf16)v.y;
      dst[2] = (__bf16)v.z; dst[3] = (__bf16)v.w;
    }
    __syncthreads();
    const v16bf a = load_afrag(As, LDA, wv * 16, 0, lane);
#pragma unroll
    for (int t = 0; t < 4; ++t) {
      const v16bf b = load_bfrag(Bs, LDB, t * 16, 0, lane);
      acc[t] = wmma_bf16(a, b, acc[t]);
    }
  }

  const int rhalf = (lane & 16) ? 8 : 0;
#pragma unroll
  for (int t = 0; t < 4; ++t) {
    const int col = nBlk + t * 16 + (lane & 15);
    const float bv = bias[col];
#pragma unroll
    for (int r = 0; r < 8; ++r) {
      const int m = mBlk + wv * 16 + rhalf + r;
      out[(size_t)m * D + col] = acc[t][r] + bv;
    }
  }
}

// ---------------------------------------------------------------------------
extern "C" void kernel_launch(void* const* d_in, const int* in_sizes, int n_in,
                              void* d_out, int out_size, void* d_ws,
                              size_t ws_size, hipStream_t stream) {
  const float* x     = (const float*)d_in[0];
  const float* qkv_w = (const float*)d_in[1];
  const float* qkv_b = (const float*)d_in[2];
  const float* out_w = (const float*)d_in[3];
  const float* out_b = (const float*)d_in[4];

  char* ws = (char*)d_ws;
  __bf16* Qn  = (__bf16*)(ws + (size_t)0);
  __bf16* Kn  = (__bf16*)(ws + (size_t)8 * 1024 * 1024);
  __bf16* Vb  = (__bf16*)(ws + (size_t)16 * 1024 * 1024);
  float*  att = (float*) (ws + (size_t)24 * 1024 * 1024);

  qkv_kernel<<<dim3(NROW / 128, (3 * D) / 64), 256, 0, stream>>>(
      x, qkv_w, qkv_b, Qn, Kn, Vb);
  attn_kernel<<<dim3(T / 128, H, BB), 256, 0, stream>>>(Qn, Kn, Vb, att);
  proj_kernel<<<dim3(NROW / 128, D / 64), 256, 0, stream>>>(
      att, out_w, out_b, (float*)d_out);
}
